// MultiheadAttention_33191507264006
// MI455X (gfx1250) — matrix operand order repaired and checked by reading, not compiled, not run on hardware
//
#include <hip/hip_runtime.h>
#include <hip/hip_bf16.h>

typedef _Float16 hf;
typedef __attribute__((ext_vector_type(16))) _Float16 v16h;
typedef __attribute__((ext_vector_type(8)))  _Float16 v8h;
typedef __attribute__((ext_vector_type(8)))  float    v8f;

static constexpr int Bb = 4;
static constexpr int Tt = 2048;
static constexpr int Ee = 1024;
static constexpr int Hh = 16;
static constexpr int HD = 64;

__device__ __forceinline__ v8h ld8(const hf* p) { return *(const v8h*)p; }
__device__ __forceinline__ v16h cat8(v8h a, v8h b) {
    return __builtin_shufflevector(a, b, 0,1,2,3,4,5,6,7,8,9,10,11,12,13,14,15);
}
__device__ __forceinline__ v8f wmma16(v16h a, v16h b, v8f c) {
    return __builtin_amdgcn_wmma_f32_16x16x32_f16(false, a, false, b, (short)0, c, false, false);
}

// ---- gfx1250 async global->LDS copy (ASYNCcnt path) via inline asm ----
// GLOBAL_LOAD_ASYNC_TO_LDS_B128, GV addressing: vdst = LDS byte address (low 32
// bits of the generic pointer, ISA 10.2 aperture rule), vaddr = 64-bit global.
__device__ __forceinline__ void cp16_async(const hf* g, hf* l) {
#if defined(__gfx1250__)
    asm volatile("global_load_async_to_lds_b128 %0, %1, off"
                 :
                 : "v"((unsigned)(uintptr_t)l), "v"((const void*)g)
                 : "memory");
#else
    *(v8h*)l = ld8(g);
#endif
}
__device__ __forceinline__ void async_drain() {
#if defined(__gfx1250__)
    asm volatile("s_wait_asynccnt 0" ::: "memory");
#endif
}

// ---------------- Kernel 1: precision conversion ----------------
__global__ __launch_bounds__(256)
void mha_convert_kernel(const float* __restrict__ x, const float* __restrict__ qkvw,
                        const float* __restrict__ projw, const unsigned char* __restrict__ mask,
                        hf* __restrict__ xh, hf* __restrict__ wqkvh, hf* __restrict__ wprojh,
                        float* __restrict__ maskf)
{
    const long n1 = (long)Bb*Tt*Ee;        // x
    const long n2 = (long)3*Ee*Ee;         // qkv_w
    const long n3 = (long)Ee*Ee;           // proj_w
    const long n4 = (long)Bb*Tt;           // mask
    long i = (long)blockIdx.x * blockDim.x + threadIdx.x;
    if (i < n1)                      { xh[i]            = (hf)x[i]; }
    else if (i < n1+n2)              { wqkvh[i-n1]      = (hf)qkvw[i-n1]; }
    else if (i < n1+n2+n3)           { wprojh[i-n1-n2]  = (hf)projw[i-n1-n2]; }
    else if (i < n1+n2+n3+n4)        { long j=i-n1-n2-n3; maskf[j] = mask[j] ? -1e30f : 0.0f; }
}

// ---------------- Kernel 2/4: tiled WMMA GEMM  out[M,N] = A[M,K] * W[N,K]^T ----------------
// Block tile 128x64, wave tile 32x32 (2x2 WMMA), K-step 32.
// mode 0: QKV epilogue (scatter Q,K,Vt as f16), mode 1: proj epilogue (f32 out + bias)
__global__ __launch_bounds__(256)
void mha_gemm_kernel(const hf* __restrict__ A, const hf* __restrict__ W,
                     int Kd, int N, int mode, const float* __restrict__ bias,
                     hf* __restrict__ qout, hf* __restrict__ kout, hf* __restrict__ vtout,
                     float* __restrict__ fout)
{
    __shared__ __align__(16) hf As[128*40];
    __shared__ __align__(16) hf Bs[64*40];
    const int tid   = threadIdx.x;
    const int mBase = blockIdx.y * 128;
    const int nBase = blockIdx.x * 64;
    const int wave  = tid >> 5, lane = tid & 31;
    const int wm    = wave & 3;        // 4 m-blocks of 32 rows
    const int wn    = wave >> 2;       // 2 n-blocks of 32 cols
    const int l16   = lane & 15;
    const bool hihalf = lane >= 16;
    const int ldr = tid >> 2;          // 0..63
    const int ldc = (tid & 3) * 8;     // 0,8,16,24

    v8f acc[2][2] = {};

    for (int k0 = 0; k0 < Kd; k0 += 32) {
        __syncthreads();
        const hf* ga0 = &A[(size_t)(mBase +      ldr)*Kd + k0 + ldc];
        const hf* ga1 = &A[(size_t)(mBase + 64 + ldr)*Kd + k0 + ldc];
        const hf* gb  = &W[(size_t)(nBase +      ldr)*Kd + k0 + ldc];
        cp16_async(ga0, &As[ ldr      *40 + ldc]);
        cp16_async(ga1, &As[(64 + ldr)*40 + ldc]);
        cp16_async(gb,  &Bs[ ldr      *40 + ldc]);
        async_drain();
        __syncthreads();
        if (k0 + 32 < Kd) {
            __builtin_prefetch(ga0 + 32, 0, 1);
            __builtin_prefetch(ga1 + 32, 0, 1);
            __builtin_prefetch(gb  + 32, 0, 1);
        }
        // A fragments 16x32: lanes<16 K {0..7,16..23}; lanes>=16 K {8..15,24..31}
        const int ak = hihalf ? 8 : 0;
        v16h af[2];
        #pragma unroll
        for (int i = 0; i < 2; ++i) {
            const int ar = wm*32 + i*16 + l16;
            af[i] = cat8(ld8(&As[ar*40 + ak]), ld8(&As[ar*40 + ak + 16]));
        }
        // B fragments 32x16: lanes<16 K {0..7,16..23}; lanes>=16 K {8..15,24..31}
        const int bk = hihalf ? 8 : 0;
        v16h bf[2];
        #pragma unroll
        for (int j = 0; j < 2; ++j) {
            const int br = wn*32 + j*16 + l16;
            bf[j] = cat8(ld8(&Bs[br*40 + bk]), ld8(&Bs[br*40 + bk + 16]));
        }
        #pragma unroll
        for (int i = 0; i < 2; ++i)
            #pragma unroll
            for (int j = 0; j < 2; ++j)
                acc[i][j] = wmma16(af[i], bf[j], acc[i][j]);
    }

    #pragma unroll
    for (int i = 0; i < 2; ++i) {
        const int mrow0 = mBase + wm*32 + i*16 + (hihalf ? 8 : 0);
        #pragma unroll
        for (int j = 0; j < 2; ++j) {
            const int n  = nBase + wn*32 + j*16 + l16;
            const float bv = bias[n];
            #pragma unroll
            for (int r = 0; r < 8; ++r) {
                const int m  = mrow0 + r;
                const float val = acc[i][j][r] + bv;
                if (mode == 0) {
                    const int b = m / Tt, t = m % Tt;
                    const int sect = n >> 10;          // n / E
                    const int nn   = n & (Ee-1);
                    const int h    = nn >> 6, d = nn & (HD-1);
                    const size_t qi = ((size_t)(b*Hh + h)*Tt + t)*HD + d;
                    if (sect == 0)      qout[qi] = (hf)(val * 0.125f);  // pre-scale Q by 1/sqrt(HD)
                    else if (sect == 1) kout[qi] = (hf)val;
                    else vtout[((size_t)(b*Hh + h)*HD + d)*Tt + t] = (hf)val;  // V transposed
                } else {
                    fout[(size_t)m*N + n] = val;
                }
            }
        }
    }
}

// ---------------- Kernel 3: flash attention, one wave = 16 query rows ----------------
__global__ __launch_bounds__(256)
void mha_flash_kernel(const hf* __restrict__ Q, const hf* __restrict__ Kk,
                      const hf* __restrict__ Vt, const float* __restrict__ maskf,
                      hf* __restrict__ ctx)
{
    const int tid  = threadIdx.x;
    const int lane = tid & 31;
    const int gw   = blockIdx.x * 8 + (tid >> 5);
    const int qtile = gw % (Tt/16);
    const int hh    = (gw / (Tt/16)) % Hh;
    const int b     = gw / ((Tt/16) * Hh);
    const int l16   = lane & 15;
    const bool hihalf = lane >= 16;
    const int qBase = qtile * 16;

    const hf* qp = Q  + ((size_t)(b*Hh + hh)*Tt + qBase) * HD;
    const hf* kp = Kk + ((size_t)(b*Hh + hh)*Tt) * HD;
    const hf* vp = Vt + ((size_t)(b*Hh + hh)*HD) * Tt;
    const float* mp = maskf + (size_t)b*Tt;

    // Q^T B-fragments (K-dim = head dim): lane = query column
    const hf* qrow = qp + l16*HD + (hihalf ? 8 : 0);
    const v16h qf0 = cat8(ld8(qrow +  0), ld8(qrow + 16));   // d 0..31
    const v16h qf1 = cat8(ld8(qrow + 32), ld8(qrow + 48));   // d 32..63

    float m_i = -1e30f, l_i = 0.0f;
    v8f o0 = {}, o1 = {}, o2 = {}, o3 = {};

    for (int kt = 0; kt < Tt; kt += 32) {
        // S^T = K * Q^T for two 16-key subtiles
        v8f s0 = {}, s1 = {};
        {
            const hf* krow = kp + (size_t)(kt + l16)*HD + (hihalf ? 8 : 0);
            v16h kf0 = cat8(ld8(krow +  0), ld8(krow + 16));
            v16h kf1 = cat8(ld8(krow + 32), ld8(krow + 48));
            s0 = wmma16(kf0, qf0, s0);
            s0 = wmma16(kf1, qf1, s0);
            krow += 16*HD;
            kf0 = cat8(ld8(krow +  0), ld8(krow + 16));
            kf1 = cat8(ld8(krow + 32), ld8(krow + 48));
            s1 = wmma16(kf0, qf0, s1);
            s1 = wmma16(kf1, qf1, s1);
        }
        // additive key mask; key index per accumulator row
        const int kb = kt + (hihalf ? 8 : 0);
        #pragma unroll
        for (int r = 0; r < 8; ++r) { s0[r] += mp[kb + r]; s1[r] += mp[kb + 16 + r]; }

        // online softmax: each lane owns one query column; combine halves via xor-16
        float tmax = -1e30f;
        #pragma unroll
        for (int r = 0; r < 8; ++r) tmax = fmaxf(tmax, fmaxf(s0[r], s1[r]));
        tmax = fmaxf(tmax, __shfl_xor(tmax, 16));
        const float m_new = fmaxf(m_i, tmax);
        const float alpha = __expf(m_i - m_new);
        float p0[8], p1[8], rs = 0.0f;
        #pragma unroll
        for (int r = 0; r < 8; ++r) {
            p0[r] = __expf(s0[r] - m_new);
            p1[r] = __expf(s1[r] - m_new);
            rs += p0[r] + p1[r];
        }
        rs += __shfl_xor(rs, 16);
        l_i = l_i * alpha + rs;
        m_i = m_new;
        o0 = o0 * alpha; o1 = o1 * alpha; o2 = o2 * alpha; o3 = o3 * alpha;

        // P^T as B operand (lane = query col, K = 32 keys): table order, no swap
        v16h pf;
        #pragma unroll
        for (int r = 0; r < 8; ++r) {
            const float q0 = __shfl_xor(p0[r], 16);
            const float q1 = __shfl_xor(p1[r], 16);
            pf[r]     = (hf)p0[r];   // keys kt+8h+r
            pf[r + 8] = (hf)p1[r];   // keys kt+16+8h+r
        }
        // O^T += V^T * P^T, four 16-row d blocks
        const hf* vrow = vp + (size_t)l16*Tt + kt + (hihalf ? 8 : 0);
        v16h vf;
        vf = cat8(ld8(vrow), ld8(vrow + 16));                 o0 = wmma16(vf, pf, o0);
        vf = cat8(ld8(vrow + 16*Tt), ld8(vrow + 16*Tt + 16)); o1 = wmma16(vf, pf, o1);
        vf = cat8(ld8(vrow + 32*Tt), ld8(vrow + 32*Tt + 16)); o2 = wmma16(vf, pf, o2);
        vf = cat8(ld8(vrow + 48*Tt), ld8(vrow + 48*Tt + 16)); o3 = wmma16(vf, pf, o3);
    }

    const float inv = 1.0f / fmaxf(l_i, 1e-20f);
    const int d0 = hihalf ? 8 : 0;
    hf* cp = ctx + ((size_t)b*Tt + qBase + l16)*Ee + hh*HD + d0;
    #pragma unroll
    for (int r = 0; r < 8; ++r) {
        cp[ 0 + r] = (hf)(o0[r] * inv);
        cp[16 + r] = (hf)(o1[r] * inv);
        cp[32 + r] = (hf)(o2[r] * inv);
        cp[48 + r] = (hf)(o3[r] * inv);
    }
}

// ---------------- host side ----------------
extern "C" void kernel_launch(void* const* d_in, const int* in_sizes, int n_in,
                              void* d_out, int out_size, void* d_ws, size_t ws_size,
                              hipStream_t stream) {
    (void)in_sizes; (void)n_in; (void)out_size; (void)ws_size;
    const float*         x      = (const float*)d_in[0];
    const unsigned char* mask   = (const unsigned char*)d_in[1];
    const float*         qkv_w  = (const float*)d_in[2];
    const float*         qkv_b  = (const float*)d_in[3];
    const float*         proj_w = (const float*)d_in[4];
    const float*         proj_b = (const float*)d_in[5];
    float*               out    = (float*)d_out;

    char* ws = (char*)d_ws;
    const size_t nBTE = (size_t)Bb*Tt*Ee;       // 8,388,608
    size_t off = 0;
    hf* xh     = (hf*)(ws + off); off += nBTE * 2;
    hf* wqkvh  = (hf*)(ws + off); off += (size_t)3*Ee*Ee * 2;
    hf* wprojh = (hf*)(ws + off); off += (size_t)Ee*Ee * 2;
    hf* qbuf   = (hf*)(ws + off); off += nBTE * 2;
    hf* kbuf   = (hf*)(ws + off); off += nBTE * 2;
    hf* vtbuf  = (hf*)(ws + off); off += nBTE * 2;
    hf* ctx    = (hf*)(ws + off); off += nBTE * 2;
    float* maskf = (float*)(ws + off); off += (size_t)Bb*Tt * 4;

    // 1) convert inputs to f16 / additive mask
    {
        const long total = (long)nBTE + 3L*Ee*Ee + (long)Ee*Ee + (long)Bb*Tt;
        const int blocks = (int)((total + 255) / 256);
        mha_convert_kernel<<<blocks, 256, 0, stream>>>(x, qkv_w, proj_w, mask,
                                                       xh, wqkvh, wprojh, maskf);
    }
    // 2) QKV projection: [B*T,E] x [3E,E]^T, scatter into Q/K/V^T head layouts
    {
        dim3 grid(3*Ee/64, Bb*Tt/128);
        mha_gemm_kernel<<<grid, 256, 0, stream>>>(xh, wqkvh, Ee, 3*Ee, 0, qkv_b,
                                                  qbuf, kbuf, vtbuf, nullptr);
    }
    // 3) flash attention -> ctx [B,T,E] f16
    {
        const int blocks = Bb*Hh*(Tt/16)/8;   // 1024
        mha_flash_kernel<<<blocks, 256, 0, stream>>>(qbuf, kbuf, vtbuf, maskf, ctx);
    }
    // 4) output projection: [B*T,E] x [E,E]^T + bias -> f32 d_out
    {
        dim3 grid(Ee/64, Bb*Tt/128);
        mha_gemm_kernel<<<grid, 256, 0, stream>>>(ctx, wprojh, Ee, Ee, 1, proj_b,
                                                  nullptr, nullptr, nullptr, out);
    }
}
